// MultiHeadAttention4DisKT_21320217657946
// MI455X (gfx1250) — matrix operand order repaired and checked by reading, not compiled, not run on hardware
//
#include <hip/hip_runtime.h>
#include <hip/hip_bf16.h>
#include <stdint.h>

#define HD __device__ __forceinline__

typedef __attribute__((ext_vector_type(16))) _Float16 v16h;
typedef __attribute__((ext_vector_type(8)))  _Float16 v8h;
typedef __attribute__((ext_vector_type(4)))  _Float16 v4h;
typedef __attribute__((ext_vector_type(8)))  float    v8f;
typedef __attribute__((ext_vector_type(4)))  float    v4f;

constexpr int B_  = 128;
constexpr int S_  = 200;
constexpr int D_  = 512;
constexpr int H_  = 8;
constexpr int DF_ = 64;
constexpr int S2  = 224;           // S padded to multiple of 32
constexpr int BS  = B_ * S_;       // 25600
constexpr float SCALE = 0.125f;    // 1/sqrt(64)

// ---------------- WMMA wrapper ----------------
HD v8f wmma_f16(v16h a, v16h b, v8f c) {
  return __builtin_amdgcn_wmma_f32_16x16x32_f16(
      /*neg_a=*/false, a, /*neg_b=*/false, b,
      /*c_mod=*/(short)0, c, /*reuse_a=*/false, /*reuse_b=*/false);
}

HD v8f vzero8() {
  v8f z;
#pragma unroll
  for (int e = 0; e < 8; ++e) z[e] = 0.0f;
  return z;
}

// ---------------- fragment loaders (f16 only now) ----------------
// A-matrix 16x32 f16 (MxK), contiguous K order:
//   lanes 0-15: row=lane, elems 0..15 -> K=k0+0..15
//   lanes16-31: row=lane-16, elems 0..15 -> K=k0+16..31
HD v16h load_a_f16(const _Float16* p, int lda, int row0, int k0, int lane) {
  const int r  = row0 + (lane & 15);
  const int kb = k0 + ((lane >> 4) << 4);
  const _Float16* base = p + (long)r * lda + kb;
  v8h lo = *(const v8h*)(base);
  v8h hi = *(const v8h*)(base + 8);
  v16h a;
#pragma unroll
  for (int e = 0; e < 8; ++e) { a[e] = lo[e]; a[8 + e] = hi[e]; }
  return a;
}

// B-matrix 32x16 f16 (KxN), loaded from B-transposed storage BT[N][K]:
//   lanes 0-15: col N = n0+lane, elems 0..15 -> K=k0+0..15
//   lanes16-31: col N = n0+lane-16, elems -> K=k0+16..31
HD v16h load_bt_f16(const _Float16* p, int ldk, int n0, int k0, int lane) {
  const int n  = n0 + (lane & 15);
  const int kb = k0 + ((lane >> 4) << 4);
  const _Float16* base = p + (long)n * ldk + kb;
  v8h lo = *(const v8h*)(base);
  v8h hi = *(const v8h*)(base + 8);
  v16h b;
#pragma unroll
  for (int e = 0; e < 8; ++e) { b[e] = lo[e]; b[8 + e] = hi[e]; }
  return b;
}

// ---------------- utility kernels ----------------
__global__ void fill_zero_u32(uint32_t* p, long n) {
  long i = (long)blockIdx.x * blockDim.x + threadIdx.x;
  const long stride = (long)gridDim.x * blockDim.x;
  for (; i < n; i += stride) p[i] = 0u;
}

// one-shot fp32 -> f16 conversion (vectorized 4-wide)
__global__ void cvt_f32_to_f16(const float* __restrict__ src,
                               _Float16* __restrict__ dst, long n4) {
  long i = (long)blockIdx.x * blockDim.x + threadIdx.x;
  const long stride = (long)gridDim.x * blockDim.x;
  for (; i < n4; i += stride) {
    v4f x = *(const v4f*)(src + i * 4);
    v4h y;
#pragma unroll
    for (int e = 0; e < 4; ++e) y[e] = (_Float16)x[e];
    *(v4h*)(dst + i * 4) = y;
  }
}

// ---------------- GEMM kernels ----------------
// QKV projection: dst = X[25600,512](f16) @ W^T(f16) + bias, head-scattered.
// Block = 128 threads (4 waves), block tile 128x64, wave tile 32x64.
// vlayout==0 : dst[b][h][s][df]  (Q, K)
// vlayout==1 : dst[b][h][df][s]  (V^T)
__global__ __launch_bounds__(128) void proj_kernel(
    const _Float16* __restrict__ X, const _Float16* __restrict__ W,
    const float* __restrict__ bias, _Float16* __restrict__ dst, int vlayout) {
  const int lane = threadIdx.x & 31;
  const int wave = threadIdx.x >> 5;                // 0..3
  const int n0   = blockIdx.x * 64;                 // output feature tile
  const int m0   = blockIdx.y * 128 + wave * 32;    // 25600/128 = 200 exact

  v8f acc[2][4];
#pragma unroll
  for (int r = 0; r < 2; ++r)
#pragma unroll
    for (int j = 0; j < 4; ++j) acc[r][j] = vzero8();

  for (int k0 = 0; k0 < D_; k0 += 32) {
    // prefetch next-next A k-block (hides k-loop latency beyond clause window)
    if (k0 + 64 < D_)
      __builtin_prefetch(X + (long)(m0 + (lane & 15)) * D_ + k0 + 64, 0, 1);
    v16h a0 = load_a_f16(X, D_, m0,      k0, lane);
    v16h a1 = load_a_f16(X, D_, m0 + 16, k0, lane);
#pragma unroll
    for (int j = 0; j < 4; ++j) {
      v16h b = load_bt_f16(W, D_, n0 + j * 16, k0, lane);
      acc[0][j] = wmma_f16(a0, b, acc[0][j]);
      acc[1][j] = wmma_f16(a1, b, acc[1][j]);
    }
  }

  const int g = lane >> 4, nl = lane & 15;
#pragma unroll
  for (int j = 0; j < 4; ++j) {
    const int n  = n0 + j * 16 + nl;
    const int h  = n >> 6;
    const int df = n & 63;
    const float bn = bias[n];
#pragma unroll
    for (int r = 0; r < 2; ++r) {
#pragma unroll
      for (int i = 0; i < 8; ++i) {
        const int m = m0 + r * 16 + i + 8 * g;
        const int b = m / S_;
        const int s = m - b * S_;
        const float val = acc[r][j][i] + bn;
        long idx;
        if (vlayout)
          idx = ((long)((b * H_ + h) * DF_ + df)) * S2 + s;
        else
          idx = ((long)((b * H_ + h) * S2 + s)) * DF_ + df;
        dst[idx] = (_Float16)val;
      }
    }
  }
}

// Output projection: out = concat[25600,512](f16) @ Wo^T(f16) + bo (fp32 out)
__global__ __launch_bounds__(128) void oproj_kernel(
    const _Float16* __restrict__ A, const _Float16* __restrict__ W,
    const float* __restrict__ bias, float* __restrict__ out) {
  const int lane = threadIdx.x & 31;
  const int wave = threadIdx.x >> 5;
  const int n0   = blockIdx.x * 64;
  const int m0   = blockIdx.y * 128 + wave * 32;

  v8f acc[2][4];
#pragma unroll
  for (int r = 0; r < 2; ++r)
#pragma unroll
    for (int j = 0; j < 4; ++j) acc[r][j] = vzero8();

  for (int k0 = 0; k0 < D_; k0 += 32) {
    if (k0 + 64 < D_)
      __builtin_prefetch(A + (long)(m0 + (lane & 15)) * D_ + k0 + 64, 0, 1);
    v16h a0 = load_a_f16(A, D_, m0,      k0, lane);
    v16h a1 = load_a_f16(A, D_, m0 + 16, k0, lane);
#pragma unroll
    for (int j = 0; j < 4; ++j) {
      v16h b = load_bt_f16(W, D_, n0 + j * 16, k0, lane);
      acc[0][j] = wmma_f16(a0, b, acc[0][j]);
      acc[1][j] = wmma_f16(a1, b, acc[1][j]);
    }
  }

  const int g = lane >> 4, nl = lane & 15;
#pragma unroll
  for (int j = 0; j < 4; ++j) {
    const int n = n0 + j * 16 + nl;
    const float bn = bias[n];
#pragma unroll
    for (int r = 0; r < 2; ++r)
#pragma unroll
      for (int i = 0; i < 8; ++i) {
        const int m = m0 + r * 16 + i + 8 * g;
        out[(long)m * D_ + n] = acc[r][j][i] + bn;
      }
  }
}

// ---------------- flash attention ----------------
// One wave per (b, h, 16-query tile). Causal + col<200 mask, online softmax,
// zero_pad zeroes query row 0's attention output.
__global__ __launch_bounds__(32) void attn_kernel(
    const _Float16* __restrict__ Qh, const _Float16* __restrict__ Kh,
    const _Float16* __restrict__ Vt, const int* __restrict__ zp_ptr,
    _Float16* __restrict__ concat) {
  __shared__ _Float16 Plds[16 * 32];

  const int lane = threadIdx.x;
  const int qt = blockIdx.x;     // 0..12
  const int h  = blockIdx.y;
  const int b  = blockIdx.z;
  const int zp = *zp_ptr;

  const _Float16* Qbh = Qh + (long)(b * H_ + h) * S2 * DF_;
  const _Float16* Kbh = Kh + (long)(b * H_ + h) * S2 * DF_;
  const _Float16* Vbh = Vt + (long)(b * H_ + h) * DF_ * S2;

  const v16h qa0 = load_a_f16(Qbh, DF_, qt * 16, 0, lane);
  const v16h qa1 = load_a_f16(Qbh, DF_, qt * 16, 32, lane);

  v8f acc[4];
#pragma unroll
  for (int j = 0; j < 4; ++j) acc[j] = vzero8();
  float m_i[8], l_i[8];
#pragma unroll
  for (int i = 0; i < 8; ++i) { m_i[i] = -1e30f; l_i[i] = 0.0f; }

  const int g = lane >> 4, nl = lane & 15;
  const int nkt = (qt * 16 + 16 + 31) / 32;   // key chunks up to diagonal

  for (int kt = 0; kt < nkt; ++kt) {
    const int c0 = kt * 32;
    v8f s0 = vzero8(), s1 = vzero8();
    {
      v16h kb = load_bt_f16(Kbh, DF_, c0, 0, lane);
      s0 = wmma_f16(qa0, kb, s0);
      kb = load_bt_f16(Kbh, DF_, c0, 32, lane);
      s0 = wmma_f16(qa1, kb, s0);
      kb = load_bt_f16(Kbh, DF_, c0 + 16, 0, lane);
      s1 = wmma_f16(qa0, kb, s1);
      kb = load_bt_f16(Kbh, DF_, c0 + 16, 32, lane);
      s1 = wmma_f16(qa1, kb, s1);
    }

    __syncthreads();   // previous iteration's P readers are done
#pragma unroll
    for (int i = 0; i < 8; ++i) {
      const int row  = qt * 16 + i + 8 * g;
      const int col0 = c0 + nl;
      const int col1 = col0 + 16;
      const bool v0 = (col0 <= row) && (col0 < S_);
      const bool v1 = (col1 <= row) && (col1 < S_);
      float x0 = v0 ? s0[i] * SCALE : -1e30f;
      float x1 = v1 ? s1[i] * SCALE : -1e30f;
      float cm = fmaxf(x0, x1);
      cm = fmaxf(cm, __shfl_xor(cm, 1));
      cm = fmaxf(cm, __shfl_xor(cm, 2));
      cm = fmaxf(cm, __shfl_xor(cm, 4));
      cm = fmaxf(cm, __shfl_xor(cm, 8));   // reduce within 16-lane row group
      const float mn = fmaxf(m_i[i], cm);
      const float alpha = __expf(m_i[i] - mn);
      const float p0 = v0 ? __expf(x0 - mn) : 0.0f;
      const float p1 = v1 ? __expf(x1 - mn) : 0.0f;
      float rs = p0 + p1;
      rs += __shfl_xor(rs, 1);
      rs += __shfl_xor(rs, 2);
      rs += __shfl_xor(rs, 4);
      rs += __shfl_xor(rs, 8);
      l_i[i] = l_i[i] * alpha + rs;
      m_i[i] = mn;
#pragma unroll
      for (int j = 0; j < 4; ++j) acc[j][i] *= alpha;
      Plds[(i + 8 * g) * 32 + nl]      = (_Float16)p0;
      Plds[(i + 8 * g) * 32 + 16 + nl] = (_Float16)p1;
    }
    __syncthreads();   // P tile visible to all lanes

    const v16h pa = load_a_f16(Plds, 32, 0, 0, lane);
#pragma unroll
    for (int j = 0; j < 4; ++j) {
      v16h vb = load_bt_f16(Vbh, S2, j * 16, c0, lane);
      acc[j] = wmma_f16(pa, vb, acc[j]);
    }
  }

#pragma unroll
  for (int i = 0; i < 8; ++i) {
    const int s = qt * 16 + i + 8 * g;
    const float inv = 1.0f / l_i[i];
    const bool zero_row = (zp != 0) && (s == 0);
    if (s < S_) {
#pragma unroll
      for (int j = 0; j < 4; ++j) {
        const float val = zero_row ? 0.0f : acc[j][i] * inv;
        concat[((long)(b * S_ + s)) * D_ + h * DF_ + j * 16 + nl] = (_Float16)val;
      }
    }
  }
}

// ---------------- launcher ----------------
extern "C" void kernel_launch(void* const* d_in, const int* in_sizes, int n_in,
                              void* d_out, int out_size, void* d_ws, size_t ws_size,
                              hipStream_t stream) {
  (void)in_sizes; (void)n_in; (void)out_size; (void)ws_size;

  const float* q  = (const float*)d_in[0];
  const float* k  = (const float*)d_in[1];
  const float* v  = (const float*)d_in[2];
  const float* Wq = (const float*)d_in[3];
  const float* bq = (const float*)d_in[4];
  const float* Wk = (const float*)d_in[5];
  const float* bk = (const float*)d_in[6];
  const float* Wv = (const float*)d_in[7];
  const float* bv = (const float*)d_in[8];
  const float* Wo = (const float*)d_in[9];
  const float* bo = (const float*)d_in[10];
  // d_in[11] = mask (causal tril — implemented analytically)
  const int* zp   = (const int*)d_in[12];
  float* out = (float*)d_out;

  // workspace layout (f16 elements):
  //  Wqh|Wkh|Wvh|Woh | Xq|Xk|Xv | Qh|Kh|Vt | concat
  const long WH  = (long)D_ * D_;            //   262,144
  const long XN  = (long)BS * D_;            // 13,107,200
  const long NH  = (long)B_ * H_ * S2 * DF_; // 14,680,064
  _Float16* ws   = (_Float16*)d_ws;
  _Float16* Wqh = ws;
  _Float16* Wkh = Wqh + WH;
  _Float16* Wvh = Wkh + WH;
  _Float16* Woh = Wvh + WH;
  _Float16* Xq  = Woh + WH;
  _Float16* Xk  = Xq + XN;
  _Float16* Xv  = Xk + XN;
  _Float16* Qh  = Xv + XN;
  _Float16* Kh  = Qh + NH;
  _Float16* Vt  = Kh + NH;
  _Float16* concat = Vt + NH;

  // zero Q/K/V head buffers so S->S2 pad rows/cols are exact zeros
  fill_zero_u32<<<2048, 256, 0, stream>>>((uint32_t*)Qh, (3 * NH) / 2);

  // one-shot fp32 -> f16 conversions (weights + activations)
  cvt_f32_to_f16<<<512, 256, 0, stream>>>(Wq, Wqh, WH / 4);
  cvt_f32_to_f16<<<512, 256, 0, stream>>>(Wk, Wkh, WH / 4);
  cvt_f32_to_f16<<<512, 256, 0, stream>>>(Wv, Wvh, WH / 4);
  cvt_f32_to_f16<<<512, 256, 0, stream>>>(Wo, Woh, WH / 4);
  cvt_f32_to_f16<<<2048, 256, 0, stream>>>(q, Xq, XN / 4);
  cvt_f32_to_f16<<<2048, 256, 0, stream>>>(k, Xk, XN / 4);
  cvt_f32_to_f16<<<2048, 256, 0, stream>>>(v, Xv, XN / 4);

  dim3 pgrid(D_ / 64, BS / 128);  // (8, 200)
  proj_kernel<<<pgrid, 128, 0, stream>>>(Xq, Wqh, bq, Qh, 0);
  proj_kernel<<<pgrid, 128, 0, stream>>>(Xk, Wkh, bk, Kh, 0);
  proj_kernel<<<pgrid, 128, 0, stream>>>(Xv, Wvh, bv, Vt, 1);

  dim3 agrid((S_ + 15) / 16, H_, B_);  // (13, 8, 128)
  attn_kernel<<<agrid, 32, 0, stream>>>(Qh, Kh, Vt, zp, concat);

  oproj_kernel<<<pgrid, 128, 0, stream>>>(concat, Woh, bo, out);
}
